// DimeNet_Model_70111046140170
// MI455X (gfx1250) — matrix operand order repaired and checked by reading, not compiled, not run on hardware
//
#include <hip/hip_runtime.h>
#include <hip/hip_bf16.h>

// ---------------------------------------------------------------------------
// DimeNet forward for MI455X (gfx1250, wave32, WMMA).
// All dense matmuls run through one bf16 WMMA GEMM (v_wmma_f32_16x16x32_bf16,
// f32 accumulate). Scatter/gather (triplets, segment sums) use f32 atomics.
// ---------------------------------------------------------------------------

#define H_     128
#define NR_    6
#define NS_    7
#define NB_    4
#define INTD_  64
#define OEMB_  256
#define CUTOFF_ 5.0f

typedef __attribute__((ext_vector_type(16))) __bf16 bf16x16;
typedef __attribute__((ext_vector_type(8)))  float  f32x8;

__device__ __forceinline__ float silu_f(float v) { return v / (1.0f + __expf(-v)); }

// ---------------------------------------------------------------------------
// Generic WMMA GEMM:  Out[M,N] (epilogue) = A[M,K] (f32, optional row gather)
//                                           x Bt[N,K] (bf16, pre-transposed)
// Block = 256 threads = 8 waves; wave w owns rows [blk*128 + 16w, +16),
// grid.y tiles N by 16.  Epilogue: v = acc (+ Out if accum) (+ bias);
// act; (* mulp); (+ addp); store.
// ---------------------------------------------------------------------------
__global__ void wmma_gemm_kernel(const float* __restrict__ A,
                                 const int*   __restrict__ arows,
                                 const __bf16* __restrict__ Bt,
                                 float* __restrict__ Out,
                                 const float* __restrict__ bias,
                                 const float* __restrict__ mulp,
                                 const float* __restrict__ addp,
                                 int M, int K, int N, int act, int accum)
{
    const int lane = threadIdx.x & 31;
    const int wave = threadIdx.x >> 5;
    const int m0   = blockIdx.x * 128 + wave * 16;
    const int n0   = blockIdx.y * 16;
    if (m0 >= M) return;

    int mr = m0 + (lane & 15);
    if (mr >= M) mr = M - 1;                 // clamp (tail safety)
    const long arow = arows ? (long)arows[mr] : (long)mr;
    // A fragment base: lanes 0-15 -> K=k0..k0+15 ; lanes 16-31 -> +16
    const float* __restrict__ ap = A + arow * (long)K + ((lane >> 4) << 4);
    // B fragment base: lanes 0-15 -> K=k0..k0+15 ; lanes 16-31 -> +16
    const int nn = n0 + (lane & 15);
    const __bf16* __restrict__ bp = Bt + (long)nn * K + ((lane >> 4) << 4);

    f32x8 acc = {};
    for (int k0 = 0; k0 < K; k0 += 32) {
        bf16x16 afrag, bfrag;
        const float*  a = ap + k0;
        const __bf16* b = bp + k0;
#pragma unroll
        for (int h = 0; h < 8; ++h) afrag[h]     = (__bf16)a[h];
#pragma unroll
        for (int h = 0; h < 8; ++h) afrag[h + 8] = (__bf16)a[h + 8];
#pragma unroll
        for (int h = 0; h < 16; ++h) bfrag[h] = b[h];
        acc = __builtin_amdgcn_wmma_f32_16x16x32_bf16(
            false, afrag, false, bfrag, (short)0, acc, false, false);
    }

    const float bv    = bias ? bias[nn] : 0.0f;
    const int   mbase = m0 + ((lane >> 4) << 3);   // C/D layout: M = r + 8*laneHi
#pragma unroll
    for (int r = 0; r < 8; ++r) {
        const int row = mbase + r;
        if (row >= M) continue;
        const long o = (long)row * N + nn;
        float v = acc[r] + bv;
        if (accum)    v += Out[o];
        if (act == 1) v = silu_f(v);
        if (mulp)     v *= mulp[o];
        if (addp)     v += addp[o];
        Out[o] = v;
    }
}

// f32 [K,N] row-major  ->  bf16 [N,K] row-major (transposed for B fragments)
__global__ void convt_kernel(const float* __restrict__ W, __bf16* __restrict__ Bt,
                             int K, int N)
{
    int i = blockIdx.x * blockDim.x + threadIdx.x;
    if (i >= K * N) return;
    int n = i % N, k = i / N;
    Bt[(long)n * K + k] = (__bf16)W[i];
}

__global__ void zero_kernel(float* __restrict__ p, long n)
{
    long i = (long)blockIdx.x * blockDim.x + threadIdx.x;
    if (i < n) p[i] = 0.0f;
}

// Per-edge geometry: distance, envelope, radial Bessel basis, spherical j_l.
__global__ void geom_kernel(const float* __restrict__ ea,
                            float* __restrict__ d, float* __restrict__ rbf,
                            float* __restrict__ rad, int E_)
{
    int e = blockIdx.x * blockDim.x + threadIdx.x;
    if (e >= E_) return;
    float x = ea[e * 3 + 0], y = ea[e * 3 + 1], zv = ea[e * 3 + 2];
    float dd = sqrtf(x * x + y * y + zv * zv + 1e-12f);
    d[e] = dd;
    float xc = dd / CUTOFF_;
    float xs = fmaxf(xc, 1e-6f);
    float x2 = xs * xs;
    float x5 = x2 * x2 * xs;
    float u  = 1.0f / xs - 28.0f * x5 + 48.0f * x5 * xs - 21.0f * x5 * x2;
    float env = (xc < 1.0f) ? u : 0.0f;
    const float PI = 3.14159265358979323846f;
#pragma unroll
    for (int n = 0; n < NR_; ++n)
        rbf[(long)e * NR_ + n] = env * sinf(PI * (float)(n + 1) * xc);
    for (int l = 0; l < NS_; ++l) {
#pragma unroll
        for (int n = 0; n < NR_; ++n) {
            float arg = PI * ((float)(n + 1) + 0.5f * (float)l) * xc;
            float as  = fmaxf(arg, 1e-6f);
            float s, c;
            sincosf(as, &s, &c);
            float j0 = s / as;
            float jl = j0;
            if (l >= 1) {
                float j1  = j0 / as - c / as;
                float jm2 = j0, jm1 = j1;
                jl = j1;
                for (int ll = 2; ll <= l; ++ll) {
                    jl = (2.0f * ll - 1.0f) / as * jm1 - jm2;
                    jm2 = jm1; jm1 = jl;
                }
            }
            rad[(long)e * (NS_ * NR_) + l * NR_ + n] = env * jl;
        }
    }
}

// Per-triplet angular basis: normalized Legendre of cos(angle).
__global__ void cbf_kernel(const float* __restrict__ ea, const float* __restrict__ d,
                           const int* __restrict__ ikj, const int* __restrict__ iji,
                           float* __restrict__ cbf, int T_)
{
    int t = blockIdx.x * blockDim.x + threadIdx.x;
    if (t >= T_) return;
    int eji = iji[t], ekj = ikj[t];
    float ax = ea[eji * 3 + 0], ay = ea[eji * 3 + 1], az = ea[eji * 3 + 2];
    float bx = -ea[ekj * 3 + 0], by = -ea[ekj * 3 + 1], bz = -ea[ekj * 3 + 2];
    float ca = (ax * bx + ay * by + az * bz) / (d[eji] * d[ekj] + 1e-9f);
    ca = fminf(1.0f, fmaxf(-1.0f, ca));
    const float I4PI = 0.07957747154594767f;  // 1/(4*pi)
    float pm2 = 1.0f, pm1 = ca;
    cbf[(long)t * NS_ + 0] = sqrtf(1.0f * I4PI);
    cbf[(long)t * NS_ + 1] = sqrtf(3.0f * I4PI) * ca;
    for (int l = 2; l < NS_; ++l) {
        float p = ((2.0f * l - 1.0f) * ca * pm1 - (float)(l - 1) * pm2) / (float)l;
        cbf[(long)t * NS_ + l] = sqrtf((2.0f * l + 1.0f) * I4PI) * p;
        pm2 = pm1; pm1 = p;
    }
}

// nodeemb[i] = emb_z[z[node]][c]
__global__ void nemb_kernel(const float* __restrict__ emb_z, const int* __restrict__ z,
                            float* __restrict__ out, long total)
{
    long i = (long)blockIdx.x * blockDim.x + threadIdx.x;
    if (i >= total) return;
    int node = (int)(i >> 7), c = (int)(i & 127);
    out[i] = emb_z[(long)z[node] * H_ + c];
}

// h_rbf = silu(rbf @ We_rbf + be_rbf)   (K=6, done scalar)
__global__ void hrbf_kernel(const float* __restrict__ rbf, const float* __restrict__ W,
                            const float* __restrict__ bvec, float* __restrict__ out,
                            long total)
{
    long i = (long)blockIdx.x * blockDim.x + threadIdx.x;
    if (i >= total) return;
    int e = (int)(i >> 7), c = (int)(i & 127);
    float s = bvec[c];
#pragma unroll
    for (int j = 0; j < NR_; ++j) s += rbf[(long)e * NR_ + j] * W[j * H_ + c];
    out[i] = silu_f(s);
}

// rbf_p = (rbf @ Wi_rbf1[b]) @ Wi_rbf2[b]   (6 -> 8 -> 128, scalar)
__global__ void rbfp_kernel(const float* __restrict__ rbf, const float* __restrict__ W1,
                            const float* __restrict__ W2, float* __restrict__ out,
                            long total)
{
    long i = (long)blockIdx.x * blockDim.x + threadIdx.x;
    if (i >= total) return;
    int e = (int)(i >> 7), c = (int)(i & 127);
    float r6[NR_];
#pragma unroll
    for (int j = 0; j < NR_; ++j) r6[j] = rbf[(long)e * NR_ + j];
    float o = 0.0f;
#pragma unroll
    for (int j = 0; j < 8; ++j) {
        float u = 0.0f;
#pragma unroll
        for (int k = 0; k < NR_; ++k) u += r6[k] * W1[k * 8 + j];
        o += u * W2[j * H_ + c];
    }
    out[i] = o;
}

// Fused triplet path for one interaction block:
//   sbf[t] = rad[idx_kj[t]] * cbf[t]   (42)
//   p      = (sbf @ Wi_sbf1) @ Wi_sbf2 (42->8->64)
//   agg[idx_ji[t]] += p * down[idx_kj[t]]   (f32 atomics)
__global__ void triplet_kernel(const float* __restrict__ rad, const float* __restrict__ cbf,
                               const float* __restrict__ down,
                               const int* __restrict__ ikj, const int* __restrict__ iji,
                               const float* __restrict__ W1,   // [42,8]
                               const float* __restrict__ W2,   // [8,64]
                               float* __restrict__ agg, int T_)
{
    __shared__ float sW1[NS_ * NR_ * 8];
    __shared__ float sW2[8 * INTD_];
    __shared__ float sS[NS_ * NR_];
    __shared__ float sU[8];
    for (int i = threadIdx.x; i < NS_ * NR_ * 8; i += 64) sW1[i] = W1[i];
    for (int i = threadIdx.x; i < 8 * INTD_;     i += 64) sW2[i] = W2[i];
    __syncthreads();
    const int c = threadIdx.x;   // 0..63
    for (int t = blockIdx.x; t < T_; t += gridDim.x) {
        const int ekj = ikj[t], eji = iji[t];
        if (c < NS_ * NR_)
            sS[c] = rad[(long)ekj * (NS_ * NR_) + c] * cbf[(long)t * NS_ + c / NR_];
        __syncthreads();
        if (c < 8) {
            float u = 0.0f;
            for (int i = 0; i < NS_ * NR_; ++i) u += sS[i] * sW1[i * 8 + c];
            sU[c] = u;
        }
        __syncthreads();
        float o = 0.0f;
#pragma unroll
        for (int j = 0; j < 8; ++j) o += sU[j] * sW2[j * INTD_ + c];
        atomicAdd(&agg[(long)eji * INTD_ + c], o * down[(long)ekj * INTD_ + c]);
        __syncthreads();
    }
}

// g = (rbf @ Wo_rbf[k]) * x ; vnode[dst] += g    (segment sum over edge_dst)
__global__ void gscatter_kernel(const float* __restrict__ rbf, const float* __restrict__ Wk,
                                const float* __restrict__ x, const int* __restrict__ dst,
                                float* __restrict__ vnode, long total)
{
    long i = (long)blockIdx.x * blockDim.x + threadIdx.x;
    if (i >= total) return;
    int e = (int)(i >> 7), c = (int)(i & 127);
    float s = 0.0f;
#pragma unroll
    for (int j = 0; j < NR_; ++j) s += rbf[(long)e * NR_ + j] * Wk[j * H_ + c];
    atomicAdd(&vnode[(long)dst[e] * H_ + c], s * x[i]);
}

__global__ void pool_kernel(const float* __restrict__ P, const int* __restrict__ batch,
                            float* __restrict__ gsum, long total)
{
    long i = (long)blockIdx.x * blockDim.x + threadIdx.x;
    if (i >= total) return;
    int node = (int)(i >> 7), c = (int)(i & 127);
    atomicAdd(&gsum[(long)batch[node] * H_ + c], P[i]);
}

__global__ void cnt_kernel(const int* __restrict__ batch, float* __restrict__ gcnt, int Nn)
{
    int i = blockIdx.x * blockDim.x + threadIdx.x;
    if (i >= Nn) return;
    atomicAdd(&gcnt[batch[i]], 1.0f);
}

// Mean pool -> LayerNorm -> relu MLP (128 -> 128 -> 4). One block per graph.
__global__ void head_kernel(const float* __restrict__ gsum, const float* __restrict__ gcnt,
                            const float* __restrict__ ln_g, const float* __restrict__ ln_b,
                            const float* __restrict__ W1, const float* __restrict__ b1,
                            const float* __restrict__ W2, const float* __restrict__ b2,
                            float* __restrict__ out)
{
    const int g = blockIdx.x, c = threadIdx.x;   // 128 threads
    __shared__ float sh[H_];
    __shared__ float red[H_];
    float val = gsum[(long)g * H_ + c] / gcnt[g];
    red[c] = val; __syncthreads();
    for (int s = 64; s > 0; s >>= 1) { if (c < s) red[c] += red[c + s]; __syncthreads(); }
    float mu = red[0] / (float)H_;
    __syncthreads();
    float dv = val - mu;
    red[c] = dv * dv; __syncthreads();
    for (int s = 64; s > 0; s >>= 1) { if (c < s) red[c] += red[c + s]; __syncthreads(); }
    float var = red[0] / (float)H_;
    float gn = dv * rsqrtf(var + 1e-5f) * ln_g[c] + ln_b[c];
    __syncthreads();
    sh[c] = gn; __syncthreads();
    float hh = b1[c];
    for (int j = 0; j < H_; ++j) hh += sh[j] * W1[j * H_ + c];
    hh = fmaxf(hh, 0.0f);
    __syncthreads();
    sh[c] = hh; __syncthreads();
    if (c < 4) {
        float o = b2[c];
        for (int j = 0; j < H_; ++j) o += sh[j] * W2[j * 4 + c];
        out[g * 4 + c] = o;
    }
}

// ---------------------------------------------------------------------------
extern "C" void kernel_launch(void* const* d_in, const int* in_sizes, int n_in,
                              void* d_out, int out_size, void* d_ws, size_t ws_size,
                              hipStream_t stream)
{
    (void)n_in; (void)ws_size;
    const int*   z_       = (const int*)  d_in[0];
    const int*   edge_src = (const int*)  d_in[1];
    const int*   edge_dst = (const int*)  d_in[2];
    const int*   batch    = (const int*)  d_in[3];
    const int*   idx_kj   = (const int*)  d_in[4];
    const int*   idx_ji   = (const int*)  d_in[5];
    const float* edge_attr= (const float*)d_in[6];
    const float* emb_z    = (const float*)d_in[7];
    const float* We_rbf   = (const float*)d_in[8];
    const float* be_rbf   = (const float*)d_in[9];
    const float* We       = (const float*)d_in[10];
    const float* be       = (const float*)d_in[11];
    const float* Wi_rbf1  = (const float*)d_in[12];
    const float* Wi_rbf2  = (const float*)d_in[13];
    const float* Wi_sbf1  = (const float*)d_in[14];
    const float* Wi_sbf2  = (const float*)d_in[15];
    const float* Wi_kj    = (const float*)d_in[16];
    const float* bi_kj    = (const float*)d_in[17];
    const float* Wi_ji    = (const float*)d_in[18];
    const float* bi_ji    = (const float*)d_in[19];
    const float* Wi_down  = (const float*)d_in[20];
    const float* Wi_up    = (const float*)d_in[21];
    const float* Wi_res   = (const float*)d_in[22];
    const float* bi_res   = (const float*)d_in[23];
    const float* Wi_skip  = (const float*)d_in[24];
    const float* bi_skip  = (const float*)d_in[25];
    const float* Wo_rbf   = (const float*)d_in[26];
    const float* Wo_up    = (const float*)d_in[27];
    const float* Wo_lin   = (const float*)d_in[28];
    const float* bo_lin   = (const float*)d_in[29];
    const float* Wo_out   = (const float*)d_in[30];
    const float* ln_g     = (const float*)d_in[31];
    const float* ln_b     = (const float*)d_in[32];
    const float* W1h      = (const float*)d_in[33];
    const float* b1h      = (const float*)d_in[34];
    const float* W2h      = (const float*)d_in[35];
    const float* b2h      = (const float*)d_in[36];

    const int Nn = in_sizes[0];
    const int E_ = in_sizes[1];
    const int T_ = in_sizes[4];
    const int NG = out_size / 4;

    // ---- workspace bump allocator -----------------------------------------
    char*  base = (char*)d_ws;
    size_t off  = 0;
    auto alloc = [&](size_t bytes) -> void* {
        off = (off + 255) & ~(size_t)255;
        void* p = base + off;
        off += bytes;
        return p;
    };

    // ---- convert all GEMM weights to bf16 [N,K] (transposed) --------------
    auto conv = [&](const float* W, int K, int Nc) -> const __bf16* {
        __bf16* p = (__bf16*)alloc((size_t)K * Nc * sizeof(__bf16));
        int tot = K * Nc;
        convt_kernel<<<(tot + 255) / 256, 256, 0, stream>>>(W, p, K, Nc);
        return p;
    };

    const __bf16* WeT[3];
    for (int i = 0; i < 3; ++i) WeT[i] = conv(We + (size_t)i * H_ * H_, H_, H_);
    const __bf16 *WiJiT[NB_], *WiKjT[NB_], *WiDownT[NB_], *WiUpT[NB_], *WiSkipT[NB_];
    const __bf16 *WiResT[NB_][6];
    for (int b = 0; b < NB_; ++b) {
        WiJiT[b]   = conv(Wi_ji   + (size_t)b * H_ * H_,    H_,   H_);
        WiKjT[b]   = conv(Wi_kj   + (size_t)b * H_ * H_,    H_,   H_);
        WiDownT[b] = conv(Wi_down + (size_t)b * H_ * INTD_, H_,   INTD_);
        WiUpT[b]   = conv(Wi_up   + (size_t)b * INTD_ * H_, INTD_, H_);
        WiSkipT[b] = conv(Wi_skip + (size_t)b * H_ * H_,    H_,   H_);
        for (int r = 0; r < 6; ++r)
            WiResT[b][r] = conv(Wi_res + (size_t)(b * 6 + r) * H_ * H_, H_, H_);
    }
    const __bf16 *WoUpT[NB_ + 1], *WoLinT[NB_ + 1][3], *WoOutT[NB_ + 1];
    for (int k = 0; k <= NB_; ++k) {
        WoUpT[k]  = conv(Wo_up  + (size_t)k * H_ * OEMB_,    H_,    OEMB_);
        for (int t = 0; t < 3; ++t)
            WoLinT[k][t] = conv(Wo_lin + (size_t)(k * 3 + t) * OEMB_ * OEMB_, OEMB_, OEMB_);
        WoOutT[k] = conv(Wo_out + (size_t)k * OEMB_ * H_,    OEMB_, H_);
    }

    // ---- activation buffers ------------------------------------------------
    float* d_d    = (float*)alloc((size_t)E_ * 4);
    float* d_rbf  = (float*)alloc((size_t)E_ * NR_ * 4);
    float* d_rad  = (float*)alloc((size_t)E_ * NS_ * NR_ * 4);
    float* d_cbf  = (float*)alloc((size_t)T_ * NS_ * 4);
    float* d_nemb = (float*)alloc((size_t)Nn * H_ * 4);
    float* d_hrbf = (float*)alloc((size_t)E_ * H_ * 4);   // reused as rbf_p
    float* d_x    = (float*)alloc((size_t)E_ * H_ * 4);
    float* d_xji  = (float*)alloc((size_t)E_ * H_ * 4);
    float* d_xkj  = (float*)alloc((size_t)E_ * H_ * 4);   // reused as h2
    float* d_h    = (float*)alloc((size_t)E_ * H_ * 4);
    float* d_scr  = (float*)alloc((size_t)E_ * H_ * 4);   // down|agg, aliased t1
    float* d_down = d_scr;
    float* d_agg  = d_scr + (size_t)E_ * INTD_;
    float* d_t1   = d_scr;
    float* d_vn   = (float*)alloc((size_t)Nn * H_ * 4);
    float* d_v0   = (float*)alloc((size_t)Nn * OEMB_ * 4);
    float* d_v1   = (float*)alloc((size_t)Nn * OEMB_ * 4);
    float* d_P    = (float*)alloc((size_t)Nn * H_ * 4);
    float* d_gsum = (float*)alloc((size_t)NG * H_ * 4);
    float* d_gcnt = (float*)alloc((size_t)NG * 4);

    auto gemm = [&](const float* A, const int* arows, const __bf16* Bt, float* Out,
                    const float* bias, const float* mulp, const float* addp,
                    int M, int K, int Nc, int act, int accum) {
        dim3 grid((M + 127) / 128, Nc / 16);
        wmma_gemm_kernel<<<grid, 256, 0, stream>>>(A, arows, Bt, Out, bias, mulp,
                                                   addp, M, K, Nc, act, accum);
    };
    auto zero = [&](float* p, long n) {
        zero_kernel<<<(int)((n + 255) / 256), 256, 0, stream>>>(p, n);
    };

    // ---- geometry ----------------------------------------------------------
    geom_kernel<<<(E_ + 127) / 128, 128, 0, stream>>>(edge_attr, d_d, d_rbf, d_rad, E_);
    if (T_ > 0)
        cbf_kernel<<<(T_ + 127) / 128, 128, 0, stream>>>(edge_attr, d_d, idx_kj, idx_ji,
                                                         d_cbf, T_);
    nemb_kernel<<<(int)(((long)Nn * H_ + 255) / 256), 256, 0, stream>>>(emb_z, z_, d_nemb,
                                                                        (long)Nn * H_);
    hrbf_kernel<<<(int)(((long)E_ * H_ + 255) / 256), 256, 0, stream>>>(d_rbf, We_rbf,
                                                                        be_rbf, d_hrbf,
                                                                        (long)E_ * H_);

    // ---- embedding: x = silu([e_src|e_dst|h_rbf] @ We + be) ----------------
    gemm(d_nemb, edge_src, WeT[0], d_x, nullptr, nullptr, nullptr, E_, H_, H_, 0, 0);
    gemm(d_nemb, edge_dst, WeT[1], d_x, nullptr, nullptr, nullptr, E_, H_, H_, 0, 1);
    gemm(d_hrbf, nullptr,  WeT[2], d_x, be,      nullptr, nullptr, E_, H_, H_, 1, 1);

    zero(d_P, (long)Nn * H_);

    auto out_block = [&](int k) {
        zero(d_vn, (long)Nn * H_);
        gscatter_kernel<<<(int)(((long)E_ * H_ + 255) / 256), 256, 0, stream>>>(
            d_rbf, Wo_rbf + (size_t)k * NR_ * H_, d_x, edge_dst, d_vn, (long)E_ * H_);
        gemm(d_vn, nullptr, WoUpT[k], d_v0, nullptr, nullptr, nullptr,
             Nn, H_, OEMB_, 0, 0);
        float* vin = d_v0; float* vout = d_v1;
        for (int t = 0; t < 3; ++t) {
            gemm(vin, nullptr, WoLinT[k][t], vout, bo_lin + (size_t)(k * 3 + t) * OEMB_,
                 nullptr, nullptr, Nn, OEMB_, OEMB_, 1, 0);
            float* tmp = vin; vin = vout; vout = tmp;
        }
        gemm(vin, nullptr, WoOutT[k], d_P, nullptr, nullptr, nullptr,
             Nn, OEMB_, H_, 0, 1);   // P += v @ Wo_out
    };

    out_block(0);

    // ---- interaction blocks ------------------------------------------------
    for (int b = 0; b < NB_; ++b) {
        float* d_rbfp = d_hrbf;    // h_rbf dead after embedding
        rbfp_kernel<<<(int)(((long)E_ * H_ + 255) / 256), 256, 0, stream>>>(
            d_rbf, Wi_rbf1 + (size_t)b * NR_ * 8, Wi_rbf2 + (size_t)b * 8 * H_,
            d_rbfp, (long)E_ * H_);

        gemm(d_x, nullptr, WiJiT[b], d_xji, bi_ji + (size_t)b * H_,
             nullptr, nullptr, E_, H_, H_, 1, 0);
        gemm(d_x, nullptr, WiKjT[b], d_xkj, bi_kj + (size_t)b * H_,
             d_rbfp, nullptr, E_, H_, H_, 1, 0);                 // * rbf_p
        gemm(d_xkj, nullptr, WiDownT[b], d_down, nullptr, nullptr, nullptr,
             E_, H_, INTD_, 1, 0);

        zero(d_agg, (long)E_ * INTD_);
        if (T_ > 0) {
            int tg = T_ < 16384 ? T_ : 16384;
            triplet_kernel<<<tg, 64, 0, stream>>>(
                d_rad, d_cbf, d_down, idx_kj, idx_ji,
                Wi_sbf1 + (size_t)b * NS_ * NR_ * 8,
                Wi_sbf2 + (size_t)b * 8 * INTD_, d_agg, T_);
        }
        // h = x_ji + silu(agg @ Wi_up)
        gemm(d_agg, nullptr, WiUpT[b], d_h, nullptr, nullptr, d_xji,
             E_, INTD_, H_, 1, 0);
        // h2 = h + silu(silu(h@res0+b0)@res1+b1)
        gemm(d_h,  nullptr, WiResT[b][0], d_t1, bi_res + (size_t)(b * 6 + 0) * H_,
             nullptr, nullptr, E_, H_, H_, 1, 0);
        gemm(d_t1, nullptr, WiResT[b][1], d_xkj, bi_res + (size_t)(b * 6 + 1) * H_,
             nullptr, d_h, E_, H_, H_, 1, 0);
        // x = silu(h2 @ skip + b) + x
        gemm(d_xkj, nullptr, WiSkipT[b], d_x, bi_skip + (size_t)b * H_,
             nullptr, d_x, E_, H_, H_, 1, 0);
        for (int r = 2; r <= 4; r += 2) {
            gemm(d_x,  nullptr, WiResT[b][r],     d_t1,
                 bi_res + (size_t)(b * 6 + r) * H_,     nullptr, nullptr,
                 E_, H_, H_, 1, 0);
            gemm(d_t1, nullptr, WiResT[b][r + 1], d_x,
                 bi_res + (size_t)(b * 6 + r + 1) * H_, nullptr, d_x,
                 E_, H_, H_, 1, 0);
        }
        out_block(b + 1);
    }

    // ---- pooling + head ----------------------------------------------------
    zero(d_gsum, (long)NG * H_);
    zero(d_gcnt, NG);
    pool_kernel<<<(int)(((long)Nn * H_ + 255) / 256), 256, 0, stream>>>(
        d_P, batch, d_gsum, (long)Nn * H_);
    cnt_kernel<<<(Nn + 127) / 128, 128, 0, stream>>>(batch, d_gcnt, Nn);
    head_kernel<<<NG, 128, 0, stream>>>(d_gsum, d_gcnt, ln_g, ln_b, W1h, b1h,
                                        W2h, b2h, (float*)d_out);
}
